// Model_17274358465009
// MI455X (gfx1250) — matrix operand order repaired and checked by reading, not compiled, not run on hardware
//
#include <hip/hip_runtime.h>
#include <math.h>

// ---------------------------------------------------------------------------
// CDNA5 (gfx1250) wave32 WMMA GAT implementation.
// B weights staged in LDS in WMMA-fragment order -> ds_load_b128 fragment reads.
// ---------------------------------------------------------------------------

typedef _Float16 v16h __attribute__((ext_vector_type(16)));
typedef float    v8f  __attribute__((ext_vector_type(8)));

union HF16 { v16h v; _Float16 h[16]; };
union F8   { v8f  v; float    f[8];  };

// Order-preserving float -> uint key for atomic max (handles negatives).
__device__ __forceinline__ unsigned fkey(float f) {
    unsigned u = __float_as_uint(f);
    return (u & 0x80000000u) ? ~u : (u | 0x80000000u);
}
__device__ __forceinline__ float funkey(unsigned k) {
    return (k & 0x80000000u) ? __uint_as_float(k ^ 0x80000000u)
                             : __uint_as_float(~k);
}

// Build a 16x32 f16 A fragment (wave32 layout) from a row-major f32 row ptr.
// lane<16 covers K {0..7,16..23}+khalf0, lane>=16 covers {8..15,24..31}.
__device__ __forceinline__ v16h load_a_frag(const float* __restrict__ ap) {
    float4 f0 = *(const float4*)(ap);
    float4 f1 = *(const float4*)(ap + 4);
    float4 f2 = *(const float4*)(ap + 16);
    float4 f3 = *(const float4*)(ap + 20);
    HF16 af;
    af.h[0]=(_Float16)f0.x; af.h[1]=(_Float16)f0.y; af.h[2]=(_Float16)f0.z; af.h[3]=(_Float16)f0.w;
    af.h[4]=(_Float16)f1.x; af.h[5]=(_Float16)f1.y; af.h[6]=(_Float16)f1.z; af.h[7]=(_Float16)f1.w;
    af.h[8]=(_Float16)f2.x; af.h[9]=(_Float16)f2.y; af.h[10]=(_Float16)f2.z; af.h[11]=(_Float16)f2.w;
    af.h[12]=(_Float16)f3.x; af.h[13]=(_Float16)f3.y; af.h[14]=(_Float16)f3.z; af.h[15]=(_Float16)f3.w;
    return af.v;
}

// ---------------------------------------------------------------------------
// Zero the 400MB output matrix (float4 vectorized).
// ---------------------------------------------------------------------------
__global__ void zero_k(float* __restrict__ p, size_t total) {
    size_t gid = ((size_t)blockIdx.x * blockDim.x + threadIdx.x) * 4;
    if (gid + 3 < total) {
        float4 z = {0.f, 0.f, 0.f, 0.f};
        *(float4*)(p + gid) = z;
    } else {
        for (int i = 0; i < 4; ++i)
            if (gid + i < total) p[gid + i] = 0.f;
    }
}

// ---------------------------------------------------------------------------
// Node-feature GEMM: Out[n x ND] = A[n x KD] @ W[KD x ND] (+bias).
// One wave computes a 16 x ND row tile via v_wmma_f32_16x16x32_f16.
// W is staged into LDS pre-swizzled into per-(kstep,ntile,lane) fragments so
// each lane reads its whole B fragment as one 32-byte LDS access.
// ---------------------------------------------------------------------------
template<int KD, int ND, bool BIAS>
__global__ __launch_bounds__(128)
void gemm_node(const float* __restrict__ A, const float* __restrict__ W,
               const float* __restrict__ bias, float* __restrict__ Out, int n)
{
    constexpr int NT = ND / 16;   // 16-wide C tiles
    constexpr int KT = KD / 32;   // k-steps of 32

    __shared__ alignas(32) _Float16 Bs[KD * ND];   // fragment-ordered
    for (int idx = threadIdx.x; idx < KD * ND; idx += 128) {
        const int j  = idx & 15;            // element within lane fragment
        const int l  = (idx >> 4) & 31;     // lane
        const int f  = idx >> 9;            // fragment id = kt*NT + t
        const int kt = f / NT, t = f - kt * NT;
        const int k  = kt * 32 + ((l < 16) ? 0 : 8) + (j & 7) + ((j >> 3) << 4);
        const int nn = t * 16 + (l & 15);
        Bs[idx] = (_Float16)W[k * ND + nn];
    }
    __syncthreads();

    const int wave = threadIdx.x >> 5, lane = threadIdx.x & 31;
    const int row0 = (blockIdx.x * 4 + wave) * 16;
    if (row0 >= n) return;

    F8 acc[NT];
    #pragma unroll
    for (int t = 0; t < NT; ++t)
        #pragma unroll
        for (int j = 0; j < 8; ++j) acc[t].f[j] = 0.f;

    const int arow  = row0 + (lane & 15);
    const int arowc = (arow < n) ? arow : (n - 1);   // clamp (stores guarded)
    const int khalf = (lane >> 4) * 8;               // K sub-offset per lane half
    const int col16 = lane & 15;
    const v16h* __restrict__ Bf = (const v16h*)Bs;   // 32B fragments

    #pragma unroll
    for (int kt = 0; kt < KT; ++kt) {
        const v16h av = load_a_frag(A + (size_t)arowc * KD + kt * 32 + khalf);
        #pragma unroll
        for (int t = 0; t < NT; ++t) {
            const v16h bv = Bf[(kt * NT + t) * 32 + lane];
            acc[t].v = __builtin_amdgcn_wmma_f32_16x16x32_f16(
                false, av, false, bv, (short)0, acc[t].v, false, false);
        }
    }

    const int rbase = (lane >> 4) * 8;   // C layout: M = r + 8*(lane>=16)
    #pragma unroll
    for (int t = 0; t < NT; ++t) {
        #pragma unroll
        for (int r = 0; r < 8; ++r) {
            int row = row0 + rbase + r;
            if (row < n) {
                int col = t * 16 + col16;
                float v = acc[t].f[r];
                if (BIAS) v += bias[col];
                Out[(size_t)row * ND + col] = v;
            }
        }
    }
}

// ---------------------------------------------------------------------------
// Per-node attention dots: a_s[i] = h2[i,:]·att_src, a_d[i] = h2[i,:]·att_dst.
// ---------------------------------------------------------------------------
__global__ void att_dots_k(const float* __restrict__ h2,
                           const float* __restrict__ aws,
                           const float* __restrict__ awd,
                           float* __restrict__ a_s, float* __restrict__ a_d, int n)
{
    int node = blockIdx.x * 256 + threadIdx.x;
    if (node >= n) return;
    const float* hp = h2 + (size_t)node * 64;
    float ss = 0.f, dd = 0.f;
    #pragma unroll
    for (int k = 0; k < 64; k += 4) {
        float4 h  = *(const float4*)(hp + k);
        float4 ws = *(const float4*)(aws + k);
        float4 wd = *(const float4*)(awd + k);
        ss += h.x*ws.x + h.y*ws.y + h.z*ws.z + h.w*ws.w;
        dd += h.x*wd.x + h.y*wd.y + h.z*wd.z + h.w*wd.w;
    }
    a_s[node] = ss; a_d[node] = dd;
}

// ---------------------------------------------------------------------------
// Per-layer init: h_out = bias (broadcast), segment max keys = key(-inf)=0, s = 0.
// ---------------------------------------------------------------------------
__global__ void layer_init_k(float* __restrict__ hout, const float* __restrict__ bias,
                             unsigned* __restrict__ mkey, float* __restrict__ ssum, int n)
{
    int gid = blockIdx.x * 256 + threadIdx.x;
    if (gid < n * 64) hout[gid] = bias[gid & 63];
    if (gid < n) { mkey[gid] = 0u; ssum[gid] = 0.f; }
}

// ---------------------------------------------------------------------------
// Edge phase A: alpha = leaky_relu(a_s[src]+a_d[dst]); atomic segment max.
// Edges 0..e-1 from edge_index, e..e+n-1 are self loops.
// ---------------------------------------------------------------------------
__global__ void edge_alpha_k(const float* __restrict__ a_s, const float* __restrict__ a_d,
                             const int* __restrict__ ei, int e, int ec,
                             float* __restrict__ alpha, unsigned* __restrict__ mkey)
{
    int gid = blockIdx.x * 256 + threadIdx.x;
    if (gid >= ec) return;
    int s, d;
    if (gid < e) { s = ei[gid]; d = ei[e + gid]; } else { s = d = gid - e; }
    float al = a_s[s] + a_d[d];
    al = (al > 0.f) ? al : 0.2f * al;
    alpha[gid] = al;
    atomicMax(&mkey[d], fkey(al));
}

// ---------------------------------------------------------------------------
// Edge phase B: ex = exp(alpha - max[dst]) (in place); atomic segment sum.
// ---------------------------------------------------------------------------
__global__ void edge_exp_k(const int* __restrict__ ei, int e, int ec,
                           const unsigned* __restrict__ mkey, float* __restrict__ ssum,
                           float* __restrict__ alpha)
{
    int gid = blockIdx.x * 256 + threadIdx.x;
    if (gid >= ec) return;
    int d = (gid < e) ? ei[e + gid] : (gid - e);
    float ex = __expf(alpha[gid] - funkey(mkey[d]));
    alpha[gid] = ex;
    atomicAdd(&ssum[d], ex);
}

// ---------------------------------------------------------------------------
// Edge aggregation: h_out[dst] += h2[src] * ex/(sum[dst]+eps).
// 16 lanes per edge, float4 per lane (64 features). Prefetch the gathered row.
// ---------------------------------------------------------------------------
__global__ void edge_agg_k(const float* __restrict__ h2, const float* __restrict__ exv,
                           const float* __restrict__ ssum, const int* __restrict__ ei,
                           int e, int ec, float* __restrict__ hout)
{
    int gid  = blockIdx.x * 256 + threadIdx.x;
    int eidx = gid >> 4;
    int l    = gid & 15;
    if (eidx >= ec) return;
    int s, d;
    if (eidx < e) { s = ei[eidx]; d = ei[e + eidx]; } else { s = d = eidx - e; }
    const float* hp = h2 + (size_t)s * 64 + l * 4;
    __builtin_prefetch(hp, 0, 0);                 // global_prefetch_b8
    float a = exv[eidx] / (ssum[d] + 1e-16f);
    float4 hv = *(const float4*)(hp);
    float* op = hout + (size_t)d * 64 + l * 4;
    atomicAdd(op + 0, hv.x * a);
    atomicAdd(op + 1, hv.y * a);
    atomicAdd(op + 2, hv.z * a);
    atomicAdd(op + 3, hv.w * a);
}

// ---------------------------------------------------------------------------
// Elementwise activation (relu between layers, sigmoid after layer 5).
// ---------------------------------------------------------------------------
template<bool SIG>
__global__ void act_k(float* __restrict__ h, int total)
{
    int gid = blockIdx.x * 256 + threadIdx.x;
    if (gid >= total) return;
    float v = h[gid];
    h[gid] = SIG ? 1.f / (1.f + __expf(-v)) : (v > 0.f ? v : 0.f);
}

// ---------------------------------------------------------------------------
// Fused edge MLP + scatter: per 16-edge tile,
//   A[16x128] = concat(h[us], h[vs]) gathered, WMMA @ mlp_w1[128x32],
//   relu(+b1) -> LDS, per-edge 32-dot with mlp_w2 (+b2), atomicAdd into E.
// mlp_w1 staged in LDS in fragment order (KT=4, NT=2).
// ---------------------------------------------------------------------------
__global__ __launch_bounds__(128)
void edge_mlp_scatter(const float* __restrict__ hf, const int* __restrict__ ei,
                      const float* __restrict__ w1, const float* __restrict__ b1,
                      const float* __restrict__ w2, const float* __restrict__ b2,
                      float* __restrict__ E, int n, int e)
{
    constexpr int KD = 128, ND = 32, NT = 2, KT = 4;
    __shared__ alignas(32) _Float16 Bs[KD * ND];   // fragment-ordered
    __shared__ float tileOut[4][16][32];
    for (int idx = threadIdx.x; idx < KD * ND; idx += 128) {
        const int j  = idx & 15;
        const int l  = (idx >> 4) & 31;
        const int f  = idx >> 9;
        const int kt = f / NT, t = f - kt * NT;
        const int k  = kt * 32 + ((l < 16) ? 0 : 8) + (j & 7) + ((j >> 3) << 4);
        const int nn = t * 16 + (l & 15);
        Bs[idx] = (_Float16)w1[k * ND + nn];
    }
    __syncthreads();

    const int wave = threadIdx.x >> 5, lane = threadIdx.x & 31;
    const int e0 = (blockIdx.x * 4 + wave) * 16;
    if (e0 >= e) return;

    F8 acc[NT];
    #pragma unroll
    for (int t = 0; t < NT; ++t)
        #pragma unroll
        for (int j = 0; j < 8; ++j) acc[t].f[j] = 0.f;

    int erow = e0 + (lane & 15);
    if (erow >= e) erow = e - 1;
    const int us = ei[erow], vs = ei[e + erow];
    const int khalf = (lane >> 4) * 8;
    const int col16 = lane & 15;
    const v16h* __restrict__ Bf = (const v16h*)Bs;

    #pragma unroll
    for (int kt = 0; kt < KT; ++kt) {                // K = 128 = 4 x 32
        const int node = (kt < 2) ? us : vs;         // concat(h[us], h[vs])
        const v16h av = load_a_frag(hf + (size_t)node * 64 + (kt & 1) * 32 + khalf);
        #pragma unroll
        for (int t = 0; t < NT; ++t) {
            const v16h bv = Bf[(kt * NT + t) * 32 + lane];
            acc[t].v = __builtin_amdgcn_wmma_f32_16x16x32_f16(
                false, av, false, bv, (short)0, acc[t].v, false, false);
        }
    }

    const int rbase = (lane >> 4) * 8;
    #pragma unroll
    for (int t = 0; t < NT; ++t)
        #pragma unroll
        for (int r = 0; r < 8; ++r) {
            float v = acc[t].f[r] + b1[t * 16 + col16];
            tileOut[wave][rbase + r][t * 16 + col16] = (v > 0.f) ? v : 0.f;
        }

    // Wave-local LDS (DS ops are in-order per wave): 32 -> 1 dot + scatter.
    if (lane < 16) {
        int eidx = e0 + lane;
        if (eidx < e) {
            float v = b2[0];
            #pragma unroll
            for (int j = 0; j < 32; ++j) v += tileOut[wave][lane][j] * w2[j];
            int u = ei[eidx], w = ei[e + eidx];
            atomicAdd(&E[(size_t)u * n + w], v);
        }
    }
}

// ---------------------------------------------------------------------------
// Host launcher.
// Input order: lin1_w, lin1_b, mlp_w1, mlp_b1, mlp_w2, mlp_b2,
//              {conv_i: w, as, ad, b} x5, x, edge_index.
// ---------------------------------------------------------------------------
extern "C" void kernel_launch(void* const* d_in, const int* in_sizes, int n_in,
                              void* d_out, int out_size, void* d_ws, size_t ws_size,
                              hipStream_t stream)
{
    const float* lin1_w = (const float*)d_in[0];
    const float* lin1_b = (const float*)d_in[1];
    const float* mlp_w1 = (const float*)d_in[2];
    const float* mlp_b1 = (const float*)d_in[3];
    const float* mlp_w2 = (const float*)d_in[4];
    const float* mlp_b2 = (const float*)d_in[5];
    const float *conv_w[5], *conv_as[5], *conv_ad[5], *conv_b[5];
    for (int i = 0; i < 5; ++i) {
        conv_w[i]  = (const float*)d_in[6 + 4 * i + 0];
        conv_as[i] = (const float*)d_in[6 + 4 * i + 1];
        conv_ad[i] = (const float*)d_in[6 + 4 * i + 2];
        conv_b[i]  = (const float*)d_in[6 + 4 * i + 3];
    }
    const float* x  = (const float*)d_in[26];
    const int*   ei = (const int*)d_in[27];
    const int n  = in_sizes[26] / 128;
    const int e  = in_sizes[27] / 2;
    const int ec = e + n;                    // + self loops

    // Workspace layout (floats): P0[n*64] | P1[n*64] | a_s[n] | a_d[n] |
    //                            mkey[n] | ssum[n] | alpha[ec]
    float*    P0    = (float*)d_ws;
    float*    P1    = P0 + (size_t)n * 64;
    float*    a_s   = P1 + (size_t)n * 64;
    float*    a_d   = a_s + n;
    unsigned* mkey  = (unsigned*)(a_d + n);
    float*    ssum  = (float*)(mkey + n);
    float*    alpha = ssum + n;
    float*    E     = (float*)d_out;

    // Zero the dense output.
    {
        size_t total = (size_t)out_size;
        unsigned blocks = (unsigned)(((total + 3) / 4 + 255) / 256);
        zero_k<<<blocks, 256, 0, stream>>>(E, total);
    }

    const unsigned gtiles = (unsigned)(((n + 15) / 16 + 3) / 4);
    const unsigned nb     = (unsigned)((n + 255) / 256);
    const unsigned n64b   = (unsigned)(((size_t)n * 64 + 255) / 256);
    const unsigned ecb    = (unsigned)((ec + 255) / 256);
    const unsigned aggb   = (unsigned)(((size_t)ec * 16 + 255) / 256);

    // lin1: h0 = x @ W + b  (n x 32)
    gemm_node<128, 32, true><<<gtiles, 128, 0, stream>>>(x, lin1_w, lin1_b, P0, n);

    for (int i = 0; i < 5; ++i) {
        if (i == 0)
            gemm_node<32, 64, false><<<gtiles, 128, 0, stream>>>(P0, conv_w[0], nullptr, P1, n);
        else
            gemm_node<64, 64, false><<<gtiles, 128, 0, stream>>>(P0, conv_w[i], nullptr, P1, n);
        att_dots_k<<<nb, 256, 0, stream>>>(P1, conv_as[i], conv_ad[i], a_s, a_d, n);
        layer_init_k<<<n64b, 256, 0, stream>>>(P0, conv_b[i], mkey, ssum, n);
        edge_alpha_k<<<ecb, 256, 0, stream>>>(a_s, a_d, ei, e, ec, alpha, mkey);
        edge_exp_k<<<ecb, 256, 0, stream>>>(ei, e, ec, mkey, ssum, alpha);
        edge_agg_k<<<aggb, 256, 0, stream>>>(P1, alpha, ssum, ei, e, ec, P0);
        if (i < 4) act_k<false><<<n64b, 256, 0, stream>>>(P0, n * 64);
        else       act_k<true ><<<n64b, 256, 0, stream>>>(P0, n * 64);
    }

    // Edge MLP on original edges + scatter into E.
    {
        unsigned blocks = (unsigned)(((e + 15) / 16 + 3) / 4);
        edge_mlp_scatter<<<blocks, 128, 0, stream>>>(P0, ei, mlp_w1, mlp_b1,
                                                     mlp_w2, mlp_b2, E, n, e);
    }
}
